// WorkingMemory_79164837200276
// MI455X (gfx1250) — matrix operand order repaired and checked by reading, not compiled, not run on hardware
//
#include <hip/hip_runtime.h>
#include <hip/hip_bf16.h>

typedef __bf16 bf16;
typedef __bf16 v16bf __attribute__((ext_vector_type(16)));
typedef float  v8f   __attribute__((ext_vector_type(8)));

#define LSEQ 2048
#define DDIM 768
#define MDIM 64
#define LOG_EPS -13.815510558f   // log(1e-6)

#define WMMA_BF16(acc, a, b) \
  (acc) = __builtin_amdgcn_wmma_f32_16x16x32_bf16(false, (a), false, (b), (short)0, (acc), false, false)

// ---------------------------------------------------------------------------
// Fragment-linear layout: each 16x32 A-tile / 32x16 B-tile occupies 512 bf16
// (1KB); lane l owns the 16 contiguous bf16 at tile*512 + l*16, holding k = 16*(l/16) + s, s = 0..15
// (contiguous k order, used by A and B alike). Loads become one aligned 32B access.
// ---------------------------------------------------------------------------

// offset (within a tile) of A element (row 0..15, k 0..31)
static __device__ inline int afrag_off(int row, int kk) {
  const int lane = (row & 15) + ((kk >> 4) << 4);
  const int pos  = kk & 15;
  return lane * 16 + pos;
}
// offset (within a tile) of B element (k 0..31, n 0..15)
static __device__ inline int bfrag_off(int kk, int n) {
  return ((((kk >> 4) << 4) + n) << 4) + (kk & 15);
}

// ---------------- one-time swizzles: f32 row-major -> bf16 fragment layout --

// A: src[rows x K] row-major; grid = (rows/16, K/32), 1 wave/tile
__global__ __launch_bounds__(32)
void swizzle_a_kernel(const float* __restrict__ src, bf16* __restrict__ dst, int K) {
  const int rt = blockIdx.x, kt = blockIdx.y;
  const int lane = threadIdx.x & 31;
  const int m = lane & 15, h = lane >> 4;
  bf16* d = dst + (((size_t)rt * gridDim.y + kt) << 9) + lane * 16;
  const float* s = src + ((size_t)rt * 16 + m) * K + kt * 32;
#pragma unroll
  for (int j = 0; j < 8; ++j) {
    const int k = (j << 1) + (h << 4);
    d[2 * j]     = (bf16)s[k];
    d[2 * j + 1] = (bf16)s[k + 1];
  }
}

// B: src[K x N] row-major; grid = (K/32, N/16), 1 wave/tile
__global__ __launch_bounds__(32)
void swizzle_b_kernel(const float* __restrict__ src, bf16* __restrict__ dst, int N) {
  const int kt = blockIdx.x, nt = blockIdx.y;
  const int lane = threadIdx.x & 31;
  const int n = lane & 15, kb = (lane >> 4) << 4;
  bf16* d = dst + (((size_t)kt * gridDim.y + nt) << 9) + lane * 16;
  const float* s = src + ((size_t)(kt * 32 + kb)) * N + nt * 16 + n;
#pragma unroll
  for (int p = 0; p < 16; ++p) d[p] = (bf16)s[(size_t)p * N];
}

// ---------------- generic WMMA GEMM on fragment-layout operands -------------
// C[L x N] = A[L x K] @ B[K x N]; one wave -> 16x64 strip.
// epi: 0=(elu+1)*M^-0.5 ->f32 ; 1=elu+1 ->f32 ; 2=clip(logsigmoid(u+bias))->f32
//      3=identity -> B-fragment bf16 (for V) ; 4=identity -> f32

__global__ __launch_bounds__(32)
void gemm_frag_kernel(const v16bf* __restrict__ Af, const v16bf* __restrict__ Bf,
                      const float* __restrict__ bias,
                      float* __restrict__ Cf, bf16* __restrict__ Cfrag,
                      int N, int K, int epi) {
  const int rt = blockIdx.y;
  const int ct0 = blockIdx.x * 4;
  const int lane = threadIdx.x & 31;
  const int KT = K >> 5, NT = N >> 4;

  const v8f vzero = {0.f, 0.f, 0.f, 0.f, 0.f, 0.f, 0.f, 0.f};
  v8f acc[4] = {vzero, vzero, vzero, vzero};

#pragma unroll 2
  for (int kt = 0; kt < KT; ++kt) {
    const v16bf a = Af[((size_t)rt * KT + kt) * 32 + lane];
#pragma unroll
    for (int t = 0; t < 4; ++t) {
      const v16bf b = Bf[((size_t)kt * NT + ct0 + t) * 32 + lane];
      WMMA_BF16(acc[t], a, b);
    }
  }

  const int hh = lane >> 4, nn = lane & 15;
#pragma unroll
  for (int t = 0; t < 4; ++t) {
    const int col = (ct0 + t) * 16 + nn;
    const float bv = bias ? bias[col] : 0.f;
#pragma unroll
    for (int r = 0; r < 8; ++r) {
      const int row = rt * 16 + r + 8 * hh;
      float u = acc[t][r] + bv;
      float y;
      if (epi == 0)      { float e = u > 0.f ? u : (__expf(u) - 1.f); y = (e + 1.f) * 0.125f; }
      else if (epi == 1) { float e = u > 0.f ? u : (__expf(u) - 1.f); y = e + 1.f; }
      else if (epi == 2) { float ls = (u < 0.f ? u : 0.f) - log1pf(__expf(-fabsf(u)));
                           y = fmaxf(ls, LOG_EPS); }
      else               { y = u; }
      if (epi == 3) {
        // scatter into B-fragment layout (k = row, n = col)
        const size_t tile = ((size_t)(row >> 5)) * NT + (col >> 4);
        Cfrag[(tile << 9) + bfrag_off(row & 31, col & 15)] = (bf16)y;
      } else {
        Cf[(size_t)row * N + col] = y;
      }
    }
  }
}

// ---------------- sequential decay scan ------------------------------------
// emits q~ as A-fragments, k~ as both B-fragments of K~^T and A-fragments of K~^T

__global__ __launch_bounds__(64)
void decay_scan_kernel(const float* __restrict__ q, const float* __restrict__ k,
                       const float* __restrict__ lf,
                       bf16* __restrict__ qA, bf16* __restrict__ kB,
                       bf16* __restrict__ kA) {
  const int m = threadIdx.x;  // one thread per M column
  float acc = 0.f;
  for (int l = 0; l < LSEQ; ++l) {
    acc += lf[l * MDIM + m];
    const float c  = fminf(20.f, fmaxf(-20.f, acc));
    const bf16 qv = (bf16)(q[l * MDIM + m] * __expf(c));
    const bf16 kv = (bf16)(k[l * MDIM + m] * __expf(-c));
    // q~ as A (row=l, k=m): tile (l/16, m/32)
    qA[((((l >> 4) << 1) + (m >> 5)) << 9) + afrag_off(l & 15, m & 31)] = qv;
    // k~^T as B (k=m, n=l): tile idx = (l/16)*2 + m/32
    kB[((((l >> 4) << 1) + (m >> 5)) << 9) + bfrag_off(m & 31, l & 15)] = kv;
    // k~^T as A (row=m, k=l): tile idx = (l/32)*4 + m/16
    kA[((((l >> 5) << 2) + (m >> 4)) << 9) + afrag_off(m & 15, l & 31)] = kv;
  }
}

// ---------------- chunked linear-attention scan ----------------------------
// blockIdx.x = D-block (64 cols); 4 waves; wave w owns row-tile w of each
// 64x64 GEMM, state tiles S[m-tile=w][*] and z-segment as WMMA accumulators.
// den_intra = P @ e0, den_inter = q~ @ (z e0^T), dz = K~^T @ e0 (ones trick).

__global__ __launch_bounds__(128)
void chunk_scan_kernel(const v16bf* __restrict__ qA, const v16bf* __restrict__ kB,
                       const v16bf* __restrict__ kA, const v16bf* __restrict__ vB,
                       float* __restrict__ hout) {
  const int db   = blockIdx.x;
  const int wave = threadIdx.x >> 5;
  const int lane = threadIdx.x & 31;
  const int hh   = lane >> 4;
  const int nn   = lane & 15;

  __shared__ v16bf sS[8 * 32];   // state as B-fragments: tile (m32*4 + dt)
  __shared__ v16bf sP[8 * 32];   // masked P as A-fragments: tile (ti*2 + s32)
  __shared__ float sZ[64];
  bf16* const sSb = (bf16*)sS;
  bf16* const sPb = (bf16*)sP;

  const v8f vzero = {0.f, 0.f, 0.f, 0.f, 0.f, 0.f, 0.f, 0.f};
  v8f accS[4] = {vzero, vzero, vzero, vzero};
  v8f accZ = vzero;

  // ones-in-column-0 B fragment (constant)
  v16bf bOnes;
#pragma unroll
  for (int i = 0; i < 16; ++i) bOnes[i] = (nn == 0) ? (bf16)1.f : (bf16)0.f;

  const int ti = wave;

  for (int c = 0; c < LSEQ / 64; ++c) {
    const int t0 = c * 64;

    // --- mirror state S (bf16 B-frags) and z (f32) into LDS: pre-update ---
#pragma unroll
    for (int dt = 0; dt < 4; ++dt)
#pragma unroll
      for (int r = 0; r < 8; ++r) {
        const int m = wave * 16 + r + 8 * hh;
        sSb[((((m >> 5) << 2) + dt) << 9) + bfrag_off(m & 31, nn)] = (bf16)accS[dt][r];
      }
    if (nn == 0) {
#pragma unroll
      for (int r = 0; r < 8; ++r) sZ[wave * 16 + r + 8 * hh] = accZ[r];
    }
    __syncthreads();

    // --- P = causal(q~ K~^T), row tile ti; store as A-fragments ---
#pragma unroll
    for (int tj = 0; tj < 4; ++tj) {
      v8f acc = vzero;
      if (tj <= ti) {
#pragma unroll
        for (int ks = 0; ks < 2; ++ks) {
          const v16bf a = qA[((size_t)((c * 4 + ti) * 2 + ks)) * 32 + lane];
          const v16bf b = kB[((size_t)((c * 4 + tj) * 2 + ks)) * 32 + lane];
          WMMA_BF16(acc, a, b);
        }
        if (tj == ti) {  // diagonal: keep s <= t
#pragma unroll
          for (int r = 0; r < 8; ++r)
            if (nn > r + 8 * hh) acc[r] = 0.f;
        }
      }
#pragma unroll
      for (int r = 0; r < 8; ++r) {
        const int s = tj * 16 + nn;
        sPb[(((ti << 1) + (s >> 5)) << 9) + afrag_off(r + 8 * hh, s & 31)] = (bf16)acc[r];
      }
    }
    __syncthreads();

    // --- denominator tile: q~ @ (z e0^T) + P @ e0 (column 0 holds den) ---
    v8f accD = vzero;
#pragma unroll
    for (int ks = 0; ks < 2; ++ks) {
      v16bf zf;
#pragma unroll
      for (int i = 0; i < 16; ++i)
        zf[i] = (nn == 0) ? (bf16)sZ[ks * 32 + hh * 16 + i] : (bf16)0.f;
      const v16bf aq = qA[((size_t)((c * 4 + ti) * 2 + ks)) * 32 + lane];
      WMMA_BF16(accD, aq, zf);
      const v16bf ap = sP[(ti * 2 + ks) * 32 + lane];
      WMMA_BF16(accD, ap, bOnes);
    }

    // --- O = q~ @ S_prev + P @ V ; divide by den ; store h ---
#pragma unroll
    for (int dt = 0; dt < 4; ++dt) {
      v8f acc = vzero;
#pragma unroll
      for (int ks = 0; ks < 2; ++ks) {
        const v16bf a1 = qA[((size_t)((c * 4 + ti) * 2 + ks)) * 32 + lane];
        const v16bf b1 = sS[(ks * 4 + dt) * 32 + lane];
        WMMA_BF16(acc, a1, b1);
        const v16bf a2 = sP[(ti * 2 + ks) * 32 + lane];
        const v16bf b2 = vB[((size_t)(c * 2 + ks) * (DDIM / 16) + db * 4 + dt) * 32 + lane];
        WMMA_BF16(acc, a2, b2);
      }
#pragma unroll
      for (int r = 0; r < 8; ++r) {
        const float den = __shfl(accD[r], lane & 16, 32) + 1e-6f;
        const int tl = ti * 16 + r + 8 * hh;
        hout[(size_t)(t0 + tl) * DDIM + db * 64 + dt * 16 + nn] = acc[r] / den;
      }
    }

    // --- state update: S(m-tile=wave) += K~^T @ V ; z += K~^T @ e0 ---
#pragma unroll
    for (int ks = 0; ks < 2; ++ks) {
      const v16bf ak = kA[((size_t)(c * 2 + ks) * 4 + wave) * 32 + lane];
#pragma unroll
      for (int dt = 0; dt < 4; ++dt) {
        const v16bf b = vB[((size_t)(c * 2 + ks) * (DDIM / 16) + db * 4 + dt) * 32 + lane];
        WMMA_BF16(accS[dt], ak, b);
      }
      WMMA_BF16(accZ, ak, bOnes);
    }
    __syncthreads();
  }
}

// ---------------- LayerNorm (block per row, D = 768 = 3*256) ---------------

__global__ __launch_bounds__(256)
void layernorm_kernel(const float* __restrict__ h, const float* __restrict__ gamma,
                      const float* __restrict__ beta, float* __restrict__ out) {
  const int row = blockIdx.x;
  const int tid = threadIdx.x;
  __shared__ float red[256];

  float v0 = h[(size_t)row * DDIM + tid];
  float v1 = h[(size_t)row * DDIM + 256 + tid];
  float v2 = h[(size_t)row * DDIM + 512 + tid];

  red[tid] = v0 + v1 + v2;
  __syncthreads();
  for (int off = 128; off > 0; off >>= 1) {
    if (tid < off) red[tid] += red[tid + off];
    __syncthreads();
  }
  const float mu = red[0] * (1.f / DDIM);
  __syncthreads();

  float d0 = v0 - mu, d1 = v1 - mu, d2 = v2 - mu;
  red[tid] = d0 * d0 + d1 * d1 + d2 * d2;
  __syncthreads();
  for (int off = 128; off > 0; off >>= 1) {
    if (tid < off) red[tid] += red[tid + off];
    __syncthreads();
  }
  const float rstd = rsqrtf(red[0] * (1.f / DDIM) + 1e-5f);

  out[(size_t)row * DDIM + tid]       = d0 * rstd * gamma[tid] + beta[tid];
  out[(size_t)row * DDIM + 256 + tid] = d1 * rstd * gamma[tid + 256] + beta[tid + 256];
  out[(size_t)row * DDIM + 512 + tid] = d2 * rstd * gamma[tid + 512] + beta[tid + 512];
}

// ---------------- host-side orchestration ----------------------------------

extern "C" void kernel_launch(void* const* d_in, const int* in_sizes, int n_in,
                              void* d_out, int out_size, void* d_ws, size_t ws_size,
                              hipStream_t stream) {
  (void)in_sizes; (void)n_in; (void)out_size; (void)ws_size;

  const float* x     = (const float*)d_in[0];
  const float* Wq    = (const float*)d_in[1];
  const float* Wk    = (const float*)d_in[2];
  const float* Wv    = (const float*)d_in[3];
  const float* Wf    = (const float*)d_in[4];
  const float* bfv   = (const float*)d_in[5];
  const float* Wo    = (const float*)d_in[6];
  const float* gamma = (const float*)d_in[7];
  const float* beta  = (const float*)d_in[8];
  float* out = (float*)d_out;

  size_t off = 0;
  auto alloc = [&](size_t bytes) -> void* {
    void* p = (char*)d_ws + off;
    off += (bytes + 255) & ~(size_t)255;
    return p;
  };
  // fragment buffers (tiles of 512 bf16)
  bf16* xA   = (bf16*)alloc((size_t)(LSEQ / 16) * (DDIM / 32) * 512 * 2);
  bf16* WqB  = (bf16*)alloc((size_t)(DDIM / 32) * (MDIM / 16) * 512 * 2);
  bf16* WkB  = (bf16*)alloc((size_t)(DDIM / 32) * (MDIM / 16) * 512 * 2);
  bf16* WfB  = (bf16*)alloc((size_t)(DDIM / 32) * (MDIM / 16) * 512 * 2);
  bf16* WvB  = (bf16*)alloc((size_t)(DDIM / 32) * (DDIM / 16) * 512 * 2);
  bf16* WoB  = (bf16*)alloc((size_t)(DDIM / 32) * (DDIM / 16) * 512 * 2);
  bf16* qAfr = (bf16*)alloc((size_t)(LSEQ / 16) * (MDIM / 32) * 512 * 2);
  bf16* kBfr = (bf16*)alloc((size_t)(LSEQ / 16) * (MDIM / 32) * 512 * 2);
  bf16* kAfr = (bf16*)alloc((size_t)(LSEQ / 32) * (MDIM / 16) * 512 * 2);
  bf16* vBfr = (bf16*)alloc((size_t)(LSEQ / 32) * (DDIM / 16) * 512 * 2);
  bf16* nrmA = (bf16*)alloc((size_t)(LSEQ / 16) * (DDIM / 32) * 512 * 2);
  // f32 intermediates
  float* qf   = (float*)alloc((size_t)LSEQ * MDIM * 4);
  float* kf   = (float*)alloc((size_t)LSEQ * MDIM * 4);
  float* lff  = (float*)alloc((size_t)LSEQ * MDIM * 4);
  float* hbf  = (float*)alloc((size_t)LSEQ * DDIM * 4);
  float* nrmf = (float*)alloc((size_t)LSEQ * DDIM * 4);

  // one-time operand swizzles
  swizzle_a_kernel<<<dim3(LSEQ / 16, DDIM / 32), 32, 0, stream>>>(x, xA, DDIM);
  swizzle_b_kernel<<<dim3(DDIM / 32, MDIM / 16), 32, 0, stream>>>(Wq, WqB, MDIM);
  swizzle_b_kernel<<<dim3(DDIM / 32, MDIM / 16), 32, 0, stream>>>(Wk, WkB, MDIM);
  swizzle_b_kernel<<<dim3(DDIM / 32, MDIM / 16), 32, 0, stream>>>(Wf, WfB, MDIM);
  swizzle_b_kernel<<<dim3(DDIM / 32, DDIM / 16), 32, 0, stream>>>(Wv, WvB, DDIM);
  swizzle_b_kernel<<<dim3(DDIM / 32, DDIM / 16), 32, 0, stream>>>(Wo, WoB, DDIM);

  const dim3 gSmall(MDIM / 64, LSEQ / 16);
  const dim3 gBig(DDIM / 64, LSEQ / 16);

  // projections (fused activations)
  gemm_frag_kernel<<<gSmall, 32, 0, stream>>>((const v16bf*)xA, (const v16bf*)WqB,
                                              nullptr, qf, nullptr, MDIM, DDIM, 0);
  gemm_frag_kernel<<<gSmall, 32, 0, stream>>>((const v16bf*)xA, (const v16bf*)WkB,
                                              nullptr, kf, nullptr, MDIM, DDIM, 1);
  gemm_frag_kernel<<<gSmall, 32, 0, stream>>>((const v16bf*)xA, (const v16bf*)WfB,
                                              bfv, lff, nullptr, MDIM, DDIM, 2);
  gemm_frag_kernel<<<gBig, 32, 0, stream>>>((const v16bf*)xA, (const v16bf*)WvB,
                                            nullptr, nullptr, vBfr, DDIM, DDIM, 3);

  // sequential decay scan -> fragment-layout q~, k~
  decay_scan_kernel<<<1, 64, 0, stream>>>(qf, kf, lff, qAfr, kBfr, kAfr);

  // chunked scan (12 D-blocks x 4 waves)
  chunk_scan_kernel<<<DDIM / 64, 128, 0, stream>>>(
      (const v16bf*)qAfr, (const v16bf*)kBfr, (const v16bf*)kAfr,
      (const v16bf*)vBfr, hbf);

  layernorm_kernel<<<LSEQ, 256, 0, stream>>>(hbf, gamma, beta, nrmf);
  swizzle_a_kernel<<<dim3(LSEQ / 16, DDIM / 32), 32, 0, stream>>>(nrmf, nrmA, DDIM);

  gemm_frag_kernel<<<gBig, 32, 0, stream>>>((const v16bf*)nrmA, (const v16bf*)WoB,
                                            nullptr, out, nullptr, DDIM, DDIM, 4);
}
